// Attention_10118942950051
// MI455X (gfx1250) — matrix operand order repaired and checked by reading, not compiled, not run on hardware
//
#include <hip/hip_runtime.h>
#include <hip/hip_bf16.h>

// ---------------------------------------------------------------------------
// Memory-augmented MHA for MI455X (gfx1250). bf16 WMMA, transposed fragments:
// no LDS, no ds_bpermute — cross-lane work via v_permlanex16 only.
// B=2, N=2048, D=1024, H=16, dk=64, M=64, NKM=2112.
// ---------------------------------------------------------------------------

typedef __attribute__((ext_vector_type(16))) __bf16 v16bf;
typedef __attribute__((ext_vector_type(8)))  __bf16 v8bf;
typedef __attribute__((ext_vector_type(2)))  __bf16 v2bf;
typedef __attribute__((ext_vector_type(16))) float  v16f;
typedef __attribute__((ext_vector_type(8)))  float  v8f;
typedef __attribute__((ext_vector_type(8)))  unsigned v8u;
typedef __attribute__((ext_vector_type(4)))  unsigned v4u;

#define D_MODEL 1024
#define SEQ_N   2048
#define NKM     2112
#define NHEADS  16
#define DK      64
#define NMEM    64

__device__ __forceinline__ v8f wmma_bf16(v16bf a, v16bf b, v8f c) {
  return __builtin_amdgcn_wmma_f32_16x16x32_bf16(false, a, false, b, (short)0, c,
                                                 false, false);
}

// raw v_exp_f32: flush-to-zero below 2^-126 is fine for softmax weights
__device__ __forceinline__ float fast_exp2(float x) {
#if __has_builtin(__builtin_amdgcn_exp2f)
  return __builtin_amdgcn_exp2f(x);
#else
  return exp2f(x);
#endif
}

__device__ __forceinline__ float fast_rcp(float x) {
#if __has_builtin(__builtin_amdgcn_rcpf)
  return __builtin_amdgcn_rcpf(x);
#else
  return 1.0f / x;
#endif
}

__device__ __forceinline__ v16bf mk16(v8bf lo, v8bf hi) {
  return __builtin_shufflevector(lo, hi, 0, 1, 2, 3, 4, 5, 6, 7,
                                 8, 9, 10, 11, 12, 13, 14, 15);
}

// A-fragment (16x32): rowp = &A[row_of_lane][k_block]. Lane k offsets:
// hf*16 + 0..15 (contiguous, matching B fragments).
__device__ __forceinline__ v16bf load_a_row(const __bf16* rowp, int lane) {
  const __bf16* p = rowp + ((lane >> 4) << 4);
  return mk16(*(const v8bf*)p, *(const v8bf*)(p + 8));
}

// B-fragment (32x16) from [n][k] row-major: lane reads 16 contiguous k at hf*16.
__device__ __forceinline__ v16bf load_b_row(const __bf16* rowp, int lane) {
  const __bf16* p = rowp + ((lane >> 4) << 4);
  return mk16(*(const v8bf*)p, *(const v8bf*)(p + 8));
}

// Same, from fp32 source with inline convert (64B vector load + pk cvt).
__device__ __forceinline__ v16bf load_b_row_f32(const float* rowp, int lane) {
  const float* p = rowp + ((lane >> 4) << 4);
  v16f f = *(const v16f*)p;
  return __builtin_convertvector(f, v16bf);
}

// pack two floats -> bf16x2 dword (v_cvt_pk_bf16_f32)
__device__ __forceinline__ unsigned pk_bf16(float a, float b) {
  v2bf t = {(__bf16)a, (__bf16)b};
  return __builtin_bit_cast(unsigned, t);
}

// xor-16 cross-half exchange, pure VALU (v_permlanex16_b32, identity selects)
__device__ __forceinline__ unsigned permx16_u(unsigned x) {
  return (unsigned)__builtin_amdgcn_permlanex16(
      (int)x, (int)x, (int)0x76543210, (int)0xFEDCBA98, false, false);
}
__device__ __forceinline__ float permx16_f(float x) {
  return __builtin_bit_cast(float, permx16_u(__builtin_bit_cast(unsigned, x)));
}

// ---------------------------------------------------------------------------
// fp32 -> bf16 conversion (weights)
// ---------------------------------------------------------------------------
__global__ void cvt_k(const float* __restrict__ in, __bf16* __restrict__ out, int n) {
  int i = blockIdx.x * 256 + threadIdx.x;
  if (i < n) out[i] = (__bf16)in[i];
}

// ---------------------------------------------------------------------------
// Mask -> additive bias in exp2 domain: mb[b][j] = masked ? -1e30 : 0
// ---------------------------------------------------------------------------
__global__ void maskbias_k(const unsigned char* __restrict__ am, float* __restrict__ mb) {
  int i = blockIdx.x * 256 + threadIdx.x;
  if (i >= 2 * NKM) return;
  const int b = i / NKM, j = i % NKM;
  mb[i] = (j < SEQ_N && am[b * SEQ_N + j] != 0) ? -1e30f : 0.0f;
}

// ---------------------------------------------------------------------------
// GEMM (transposed fragments): C[row, col] = sum_d X[row,d]*W[col,d] + bias[col]
// A = W rows (out cols), B = X^T -> lane's C column is one X row, so the 8
// accumulator values are 8 CONSECUTIVE output columns: vectorized epilogue.
// Grid (16 colblocks x 64, 32 rowblocks x 128), 256 threads = 8 waves.
// OUT_MODE 0: f32 row-major (d_out); 1: bf16 row-major w/ per-batch row pad;
// OUT_MODE 2: bf16 scattered into V-transposed layout [B,H,64,2112].
// ---------------------------------------------------------------------------
template <int X_BF16, int OUT_MODE>
__global__ __launch_bounds__(256) void gemm_k(const void* __restrict__ Xp,
                                              const __bf16* __restrict__ W,
                                              const float* __restrict__ bias,
                                              void* __restrict__ outp, int rowPad) {
  const int lane = threadIdx.x & 31;
  const int wave = threadIdx.x >> 5;
  const int ln = lane & 15;
  const int hf = lane >> 4;
  const int rowX = blockIdx.y * 128 + wave * 16 + ln;  // X row == C column
  const int colBase = blockIdx.x * 64;

  v8f acc[4];
#pragma unroll
  for (int t = 0; t < 4; ++t)
#pragma unroll
    for (int r = 0; r < 8; ++r) acc[t][r] = 0.f;

  for (int kb = 0; kb < D_MODEL; kb += 32) {
    v16bf xb;
    if (X_BF16)
      xb = load_b_row(((const __bf16*)Xp) + (size_t)rowX * D_MODEL + kb, lane);
    else
      xb = load_b_row_f32(((const float*)Xp) + (size_t)rowX * D_MODEL + kb, lane);
#pragma unroll
    for (int t = 0; t < 4; ++t) {
      const __bf16* wrow = W + (size_t)(colBase + t * 16 + ln) * D_MODEL + kb;
      acc[t] = wmma_bf16(load_a_row(wrow, lane), xb, acc[t]);
    }
  }

  // epilogue: out row = rowX (lane-uniform); cols = colBase + t*16 + hf*8 + r
#pragma unroll
  for (int t = 0; t < 4; ++t) {
    const int c0 = colBase + t * 16 + hf * 8;
    const v8f bv = *(const v8f*)(bias + c0);
    if (OUT_MODE == 0) {
      float* o = (float*)outp + (size_t)rowX * D_MODEL + c0;
      float4 f0 = make_float4(acc[t][0] + bv[0], acc[t][1] + bv[1],
                              acc[t][2] + bv[2], acc[t][3] + bv[3]);
      float4 f1 = make_float4(acc[t][4] + bv[4], acc[t][5] + bv[5],
                              acc[t][6] + bv[6], acc[t][7] + bv[7]);
      *(float4*)o = f0;
      *((float4*)o + 1) = f1;
    } else if (OUT_MODE == 1) {
      const int orow = rowX + (rowX >> 11) * rowPad;  // rowX/2048 = batch
      v4u pd;
#pragma unroll
      for (int i = 0; i < 4; ++i)
        pd[i] = pk_bf16(acc[t][2 * i] + bv[2 * i], acc[t][2 * i + 1] + bv[2 * i + 1]);
      *(v8bf*)((__bf16*)outp + (size_t)orow * D_MODEL + c0) =
          __builtin_bit_cast(v8bf, pd);
    } else {
      const int bb = rowX >> 11, n = rowX & 2047;
#pragma unroll
      for (int r = 0; r < 8; ++r) {
        const int col = c0 + r;
        const int hh = col >> 6, d = col & 63;
        ((__bf16*)outp)[((size_t)((bb * NHEADS + hh) * DK + d)) * NKM + n] =
            (__bf16)(acc[t][r] + bv[r]);
      }
    }
  }
}

// ---------------------------------------------------------------------------
// Fill memory-slot rows of Kc and memory columns of Vt (x sqrt(dk)=sqrt(M)=8).
// ---------------------------------------------------------------------------
__global__ void fill_mem_k(const float* __restrict__ mk, const float* __restrict__ mv,
                           __bf16* __restrict__ Kc, __bf16* __restrict__ Vt) {
  int idx = blockIdx.x * 256 + threadIdx.x;  // [0, 2*64*1024)
  if (idx >= 2 * NMEM * D_MODEL) return;
  const int b = idx >> 16;
  const int i = (idx >> 10) & 63;
  const int c = idx & 1023;
  Kc[((size_t)b * NKM + SEQ_N + i) * D_MODEL + c] = (__bf16)(mk[i * D_MODEL + c] * 8.0f);
  const int h = c >> 6, d = c & 63;
  Vt[((size_t)((b * NHEADS + h) * DK + d)) * NKM + SEQ_N + i] =
      (__bf16)(mv[i * D_MODEL + c] * 8.0f);
}

// ---------------------------------------------------------------------------
// Flash attention, transposed: S^T = K*Q^T, O^T = V^T*P^T.
// Lane column == one query -> scalar m/l per lane, in-lane reductions plus
// one permlanex16; P^T B-fragment built in registers (pack+permlane+select).
// Grid (16 qblocks, 16 heads, 2 batches), 256 threads = 8 waves.
// ---------------------------------------------------------------------------
__global__ __launch_bounds__(256) void flash_k(const __bf16* __restrict__ Qp,
                                               const __bf16* __restrict__ Kc,
                                               const __bf16* __restrict__ Vt,
                                               const float* __restrict__ mbias,
                                               __bf16* __restrict__ ctx) {
  const int lane = threadIdx.x & 31;
  const int wave = threadIdx.x >> 5;
  const int ln = lane & 15;
  const int hf = lane >> 4;
  const int b = blockIdx.z, h = blockIdx.y;
  const int q0 = blockIdx.x * 128 + wave * 16;
  const int qrowi = b * SEQ_N + q0 + ln;  // this lane's query row

  // Q^T B-fragments (contraction k = head dim), loaded once: 16 contiguous.
  const __bf16* qrow = Qp + (size_t)qrowi * D_MODEL + h * DK;
  const v16bf bq0 = load_b_row(qrow, lane);
  const v16bf bq1 = load_b_row(qrow + 32, lane);

  v8f O[4];
#pragma unroll
  for (int t = 0; t < 4; ++t)
#pragma unroll
    for (int r = 0; r < 8; ++r) O[t][r] = 0.f;
  float m = -1e30f, l = 0.f;

  const __bf16* Kbase = Kc + (size_t)b * NKM * D_MODEL + h * DK;
  const __bf16* Vbase = Vt + (size_t)(b * NHEADS + h) * DK * NKM;
  const float* mbb = mbias + b * NKM;
  const float SC = 0.125f * 1.44269504088896f;  // 1/sqrt(dk) * log2(e)

  for (int j0 = 0; j0 < NKM; j0 += 32) {
    // S^T: frag g rows = keys j0+g*16+hf*8+r, cols = queries (lane-uniform)
    v8f S[2];
#pragma unroll
    for (int g = 0; g < 2; ++g) {
      const __bf16* krow = Kbase + (size_t)(j0 + g * 16 + ln) * D_MODEL;
      v8f s;
#pragma unroll
      for (int r = 0; r < 8; ++r) s[r] = 0.f;
      s = wmma_bf16(load_a_row(krow, lane), bq0, s);
      s = wmma_bf16(load_a_row(krow + 32, lane), bq1, s);
      const v8f mv = *(const v8f*)(mbb + j0 + g * 16 + hf * 8);
#pragma unroll
      for (int r = 0; r < 8; ++r) s[r] = s[r] * SC + mv[r];
      S[g] = s;
    }

    // online softmax over this lane's 16 key-scores + cross-half exchange
    float mx = S[0][0];
#pragma unroll
    for (int r = 1; r < 8; ++r) mx = fmaxf(mx, S[0][r]);
#pragma unroll
    for (int r = 0; r < 8; ++r) mx = fmaxf(mx, S[1][r]);
    mx = fmaxf(mx, permx16_f(mx));
    const float mnew = fmaxf(m, mx);
    const float alpha = fast_exp2(m - mnew);
    m = mnew;
    float ps = 0.f;
#pragma unroll
    for (int g = 0; g < 2; ++g)
#pragma unroll
      for (int r = 0; r < 8; ++r) {
        const float p = fast_exp2(S[g][r] - mnew);
        S[g][r] = p;
        ps += p;
      }
    ps += permx16_f(ps);
    l = l * alpha + ps;
#pragma unroll
    for (int t = 0; t < 4; ++t)
#pragma unroll
      for (int r = 0; r < 8; ++r) O[t][r] *= alpha;

    // P^T B-fragment in registers: lane element e -> key j0 + hf*16 + e.
    // e<8 from (hf?partner:own) frag[hf], e>=8 from (hf?own:partner) frag[hf].
    unsigned pk0[4], pk1[4], x0[4], x1[4];
#pragma unroll
    for (int i = 0; i < 4; ++i) {
      pk0[i] = pk_bf16(S[0][2 * i], S[0][2 * i + 1]);
      pk1[i] = pk_bf16(S[1][2 * i], S[1][2 * i + 1]);
    }
#pragma unroll
    for (int i = 0; i < 4; ++i) {
      x0[i] = permx16_u(pk0[i]);
      x1[i] = permx16_u(pk1[i]);
    }
    v8u pd;
#pragma unroll
    for (int i = 0; i < 4; ++i) {
      pd[i] = hf ? x1[i] : pk0[i];
      pd[4 + i] = hf ? pk1[i] : x0[i];
    }
    const v16bf pb = __builtin_bit_cast(v16bf, pd);

    // O^T += V^T * P^T : A-frag rows = head dims, contiguous key loads from Vt
#pragma unroll
    for (int t = 0; t < 4; ++t) {
      const __bf16* vrow = Vbase + (size_t)(t * 16 + ln) * NKM + j0;
      O[t] = wmma_bf16(load_a_row(vrow, lane), pb, O[t]);
    }
  }

  // normalize + packed 16B stores: O^T frag rows are consecutive head dims
  const float inv = fast_rcp(l);
  __bf16* crow = ctx + (size_t)qrowi * D_MODEL + h * DK + hf * 8;
#pragma unroll
  for (int t = 0; t < 4; ++t) {
    v4u pd;
#pragma unroll
    for (int i = 0; i < 4; ++i)
      pd[i] = pk_bf16(O[t][2 * i] * inv, O[t][2 * i + 1] * inv);
    *(v8bf*)(crow + t * 16) = __builtin_bit_cast(v8bf, pd);
  }
}

// ---------------------------------------------------------------------------
// Launch
// ---------------------------------------------------------------------------
extern "C" void kernel_launch(void* const* d_in, const int* in_sizes, int n_in,
                              void* d_out, int out_size, void* d_ws, size_t ws_size,
                              hipStream_t stream) {
  (void)in_sizes; (void)n_in; (void)out_size; (void)ws_size;
  const float* q  = (const float*)d_in[0];
  const float* k  = (const float*)d_in[1];
  const float* v  = (const float*)d_in[2];
  const unsigned char* amask = (const unsigned char*)d_in[3];
  const float* Wq = (const float*)d_in[4];
  const float* bq = (const float*)d_in[5];
  const float* Wk = (const float*)d_in[6];
  const float* bk = (const float*)d_in[7];
  const float* Wv = (const float*)d_in[8];
  const float* bv = (const float*)d_in[9];
  const float* Wo = (const float*)d_in[10];
  const float* bo = (const float*)d_in[11];
  const float* mk = (const float*)d_in[12];
  const float* mv = (const float*)d_in[13];

  char* ws = (char*)d_ws;
  __bf16* Qp  = (__bf16*)(ws + 0);         //  4096*1024 bf16 =  8,388,608 B
  __bf16* Kc  = (__bf16*)(ws + 8388608);   //  2*2112*1024    =  8,650,752 B
  __bf16* Vt  = (__bf16*)(ws + 17039360);  //  2*16*64*2112   =  8,650,752 B
  __bf16* ctx = (__bf16*)(ws + 25690112);  //  4096*1024      =  8,388,608 B
  __bf16* Wqb = (__bf16*)(ws + 34078720);  //  1024*1024      =  2,097,152 B
  __bf16* Wkb = (__bf16*)(ws + 36175872);
  __bf16* Wvb = (__bf16*)(ws + 38273024);
  __bf16* Wob = (__bf16*)(ws + 40370176);
  float*  mbias = (float*)(ws + 42467328); //  2*2112*4 = 16,896 B (tot ~40.5 MB)

  const int NW = D_MODEL * D_MODEL;
  cvt_k<<<(NW + 255) / 256, 256, 0, stream>>>(Wq, Wqb, NW);
  cvt_k<<<(NW + 255) / 256, 256, 0, stream>>>(Wk, Wkb, NW);
  cvt_k<<<(NW + 255) / 256, 256, 0, stream>>>(Wv, Wvb, NW);
  cvt_k<<<(NW + 255) / 256, 256, 0, stream>>>(Wo, Wob, NW);
  maskbias_k<<<(2 * NKM + 255) / 256, 256, 0, stream>>>(amask, mbias);

  const dim3 ggrid(16, 32);  // 16 col-blocks of 64, 32 row-blocks of 128
  gemm_k<0, 1><<<ggrid, 256, 0, stream>>>(q, Wqb, bq, Qp, 0);
  gemm_k<0, 1><<<ggrid, 256, 0, stream>>>(k, Wkb, bk, Kc, NMEM);
  gemm_k<0, 2><<<ggrid, 256, 0, stream>>>(v, Wvb, bv, Vt, 0);

  fill_mem_k<<<(2 * NMEM * D_MODEL + 255) / 256, 256, 0, stream>>>(mk, mv, Kc, Vt);

  flash_k<<<dim3(16, NHEADS, 2), 256, 0, stream>>>(Qp, Kc, Vt, mbias, ctx);

  gemm_k<1, 0><<<ggrid, 256, 0, stream>>>(ctx, Wob, bo, d_out, 0);
}
